// FuSAGNet_46377056862787
// MI455X (gfx1250) — matrix operand order repaired and checked by reading, not compiled, not run on hardware
//
#include <hip/hip_runtime.h>
#include <hip/hip_bf16.h>

// ---------------------------------------------------------------------------
// FuSAGNet forward for MI455X (gfx1250, wave32, WMMA + async global->LDS).
// Dominant cost: 4x (256x2048)@(2048x2048)^T GEMMs -> bf16 WMMA, f32 accum,
// weights double-buffered through LDS with global_load_async_to_lds_b128,
// fragments fetched with conflict-free ds_load_b128 (36-float row pitch).
// ---------------------------------------------------------------------------

typedef __attribute__((ext_vector_type(16))) __bf16 v16bf;
typedef __attribute__((ext_vector_type(8)))  float  v8f;

#define Bsz   256
#define Nn    128
#define Dd    64
#define TOPK  30
#define NWf   2048          // N*W
#define BN_NODES 32768      // B*N
#define EPSf  1e-5f

#define BLK_M 64
#define BLK_N 128
#define BLK_K 32
#define LDP   36            // LDS row pitch (floats): 16B-aligned, bank-clean

// Async copy of 16 bytes global -> LDS (per-lane addresses), ASYNCcnt-tracked.
__device__ __forceinline__ void async_copy16(void* lds_dst, const void* gsrc) {
  uint32_t lds = (uint32_t)(uintptr_t)lds_dst;   // low 32 bits = LDS offset
  asm volatile("global_load_async_to_lds_b128 %0, %1, off"
               :
               : "v"(lds), "v"(gsrc)
               : "memory");
}

__device__ __forceinline__ void cvt4(v16bf& f, int base, float4 c) {
  f[base + 0] = (__bf16)c.x; f[base + 1] = (__bf16)c.y;
  f[base + 2] = (__bf16)c.z; f[base + 3] = (__bf16)c.w;
}

// ---------------------------------------------------------------------------
// GEMM: out[M x Nout] = A[M x K] @ Wm[Nout x K]^T + bias (fp32 io, bf16 WMMA).
// Block: 256 threads / 8 waves -> 64x128 output tile; per wave 32x32 (2x2
// WMMA tiles). K staged in 32-wide double-buffered LDS chunks via async DMA.
// Fragment layouts per CDNA5 ISA 7.12.2 (two 8-wide k runs per lane):
//   A (16x32): lane(l&15)=M row; a[0..7]=row[half*8..+8), a[8..15]=row[16+half*8..+8)
//   B (32x16): lane(l&15)=N col; b[0..7]=row[half*8..+8), b[8..15]=row[16+half*8..+8)
//   C/D      : m = j + 8*(l>>4), n = l&15
// ---------------------------------------------------------------------------
__global__ __launch_bounds__(256)
void gemm_bf16_wmma(const float* __restrict__ A, const float* __restrict__ Wm,
                    const float* __restrict__ bias, float* __restrict__ out,
                    int M, int K, int Nout) {
  __shared__ __align__(16) float Asub[2][BLK_M][LDP];
  __shared__ __align__(16) float Bsub[2][BLK_N][LDP];

  const int tid  = threadIdx.x;
  const int lane = tid & 31, w = tid >> 5;
  const int half = lane >> 4, r = lane & 15;
  const int nBlkCnt = Nout / BLK_N;
  const int mBase = (blockIdx.x / nBlkCnt) * BLK_M;
  const int nBase = (blockIdx.x % nBlkCnt) * BLK_N;
  const int mo = (w & 1) * 32, no = (w >> 1) * 32;

  v8f acc[2][2];
  const v8f vz = {0.f,0.f,0.f,0.f,0.f,0.f,0.f,0.f};
  acc[0][0] = vz; acc[0][1] = vz; acc[1][0] = vz; acc[1][1] = vz;

  const int NK = K / BLK_K;

  // prologue: stage k-chunk 0 into buffer 0 (2 A + 4 B b128s per thread)
#pragma unroll
  for (int c = tid; c < (BLK_M * BLK_K) / 4; c += 256) {
    int row = c >> 3, col = (c & 7) << 2;
    async_copy16(&Asub[0][row][col], A + (size_t)(mBase + row) * K + col);
  }
#pragma unroll
  for (int c = tid; c < (BLK_N * BLK_K) / 4; c += 256) {
    int row = c >> 3, col = (c & 7) << 2;
    async_copy16(&Bsub[0][row][col], Wm + (size_t)(nBase + row) * K + col);
  }

  for (int kt = 0; kt < NK; ++kt) {
    const int buf = kt & 1;
    if (kt + 1 < NK) {
      const int k0 = (kt + 1) * BLK_K;
      const int nb = buf ^ 1;
#pragma unroll
      for (int c = tid; c < (BLK_M * BLK_K) / 4; c += 256) {
        int row = c >> 3, col = (c & 7) << 2;
        async_copy16(&Asub[nb][row][col],
                     A + (size_t)(mBase + row) * K + k0 + col);
      }
#pragma unroll
      for (int c = tid; c < (BLK_N * BLK_K) / 4; c += 256) {
        int row = c >> 3, col = (c & 7) << 2;
        async_copy16(&Bsub[nb][row][col],
                     Wm + (size_t)(nBase + row) * K + k0 + col);
      }
      // 6 newest (next buffer) stay in flight; current buffer is complete.
      asm volatile("s_wait_asynccnt 0x6" ::: "memory");
    } else {
      asm volatile("s_wait_asynccnt 0x0" ::: "memory");
    }
    __syncthreads();

    v16bf a[2], b[2];
#pragma unroll
    for (int mi = 0; mi < 2; ++mi) {
      const float4* ar4 =
          (const float4*)&Asub[buf][mo + mi * 16 + r][0];
      cvt4(a[mi], 0,  ar4[half * 2]);
      cvt4(a[mi], 4,  ar4[half * 2 + 1]);
      cvt4(a[mi], 8,  ar4[half * 2 + 4]);
      cvt4(a[mi], 12, ar4[half * 2 + 5]);
    }
#pragma unroll
    for (int ni = 0; ni < 2; ++ni) {
      const float4* br4 =
          (const float4*)&Bsub[buf][no + ni * 16 + r][0];
      cvt4(b[ni], 0,  br4[half * 2]);
      cvt4(b[ni], 4,  br4[half * 2 + 1]);
      cvt4(b[ni], 8,  br4[half * 2 + 4]);
      cvt4(b[ni], 12, br4[half * 2 + 5]);
    }
#pragma unroll
    for (int mi = 0; mi < 2; ++mi)
#pragma unroll
      for (int ni = 0; ni < 2; ++ni)
        acc[mi][ni] = __builtin_amdgcn_wmma_f32_16x16x32_bf16(
            false, a[mi], false, b[ni], (short)0, acc[mi][ni], false, false);
    __syncthreads();   // readers done before this buffer is refilled
  }

#pragma unroll
  for (int mi = 0; mi < 2; ++mi)
#pragma unroll
    for (int ni = 0; ni < 2; ++ni) {
      int n = nBase + no + ni * 16 + r;
      float bv = bias[n];
#pragma unroll
      for (int j = 0; j < 8; ++j) {
        int m = mBase + mo + mi * 16 + j + half * 8;
        out[(size_t)m * Nout + n] = acc[mi][ni][j] + bv;
      }
    }
}

// ---------------------------------------------------------------------------
// Per-column mean / biased variance, thread-per-column (coalesced).
// ---------------------------------------------------------------------------
__global__ void col_stats_gen(const float* __restrict__ x,
                              float* __restrict__ mean, float* __restrict__ var,
                              int Mrows, int Ncols) {
  int c = blockIdx.x * blockDim.x + threadIdx.x;
  if (c >= Ncols) return;
  float a = 0.f, b = 0.f;
  for (int r = 0; r < Mrows; ++r) {
    float v = x[(size_t)r * Ncols + c];
    a += v; b += v * v;
  }
  float m = a / (float)Mrows;
  mean[c] = m;
  var[c]  = b / (float)Mrows - m * m;
}

// out = sigmoid((x - m) * rsqrt(v+eps) * g + beta), elementwise, Ncols pow2
__global__ void bn_sigmoid_apply(const float* __restrict__ lin,
                                 const float* __restrict__ mean,
                                 const float* __restrict__ var,
                                 const float* __restrict__ g,
                                 const float* __restrict__ beta,
                                 float* __restrict__ out,
                                 int colMask, int total) {
  int id = blockIdx.x * blockDim.x + threadIdx.x;
  if (id >= total) return;
  int c = id & colMask;
  float nv = (lin[id] - mean[c]) * rsqrtf(var[c] + EPSf) * g[c] + beta[c];
  out[id] = 1.f / (1.f + __expf(-nv));
}

// rho_hat = sum over batch of z
__global__ void colsum(const float* __restrict__ z, float* __restrict__ out,
                       int Mrows, int Ncols) {
  int c = blockIdx.x * blockDim.x + threadIdx.x;
  if (c >= Ncols) return;
  float s = 0.f;
  for (int r = 0; r < Mrows; ++r) s += z[(size_t)r * Ncols + c];
  out[c] = s;
}

// ---------------------------------------------------------------------------
// 3-layer bidirectional GRU on a length-1 sequence, h0=0 (W_hh only enters
// through b_hh, exactly as the reference). One block (96 threads) per node.
// ---------------------------------------------------------------------------
__global__ __launch_bounds__(96)
void gru_embed(const float* __restrict__ emb_tables,
               const float* __restrict__ Wih, const float* __restrict__ bih,
               const float* __restrict__ bhh, float* __restrict__ embeds) {
  __shared__ float xb[64], ob[64], gb[96];
  const int node = blockIdx.x;
  const int t = threadIdx.x;
  if (t < 64) xb[t] = emb_tables[(size_t)node * 64 + t];
  __syncthreads();
  const int p = node >> 5;
  for (int l = 0; l < 3; ++l) {
    for (int d = 0; d < 2; ++d) {
      const int base = ((p * 3 + l) * 2 + d);
      const float* wrow = Wih + ((size_t)base * 96 + t) * 64;
      float gi = bih[base * 96 + t];
      for (int k = 0; k < 64; ++k) gi += wrow[k] * xb[k];
      gb[t] = gi;
      __syncthreads();
      if (t < 32) {
        const float* bh = bhh + base * 96;
        float r = 1.f / (1.f + __expf(-(gb[t]      + bh[t])));
        float z = 1.f / (1.f + __expf(-(gb[32 + t] + bh[32 + t])));
        float n = tanhf(gb[64 + t] + r * bh[64 + t]);
        ob[d * 32 + t] = (1.f - z) * n;
      }
      __syncthreads();
    }
    if (t < 64) xb[t] = ob[t];
    __syncthreads();
  }
  if (t < 64) embeds[(size_t)node * 64 + t] = xb[t];
}

// all_emb = tile(embeds, (B,1))
__global__ void write_allemb(const float* __restrict__ embeds,
                             float* __restrict__ out) {
  int id = blockIdx.x * blockDim.x + threadIdx.x;
  if (id >= BN_NODES * Dd) return;
  int n = id >> 6, d = id & 63;
  out[id] = embeds[((n & 127) << 6) + d];
}

// Top-k cosine similarity graph. One block (128 threads) per row.
__global__ __launch_bounds__(128)
void topk_graph(const float* __restrict__ embeds, int* __restrict__ idx) {
  __shared__ float sims[128];
  const int i = blockIdx.x, t = threadIdx.x;
  float dot = 0.f, ni = 0.f, nj = 0.f;
  for (int k = 0; k < 64; ++k) {
    float a = embeds[i * 64 + k], b = embeds[t * 64 + k];
    dot += a * b; ni += a * a; nj += b * b;
  }
  sims[t] = dot * rsqrtf(ni) * rsqrtf(nj);
  __syncthreads();
  if (t == 0) {
    for (int k = 0; k < TOPK; ++k) {
      int best = 0; float bv = -1e30f;
      for (int j = 0; j < 128; ++j)
        if (sims[j] > bv) { bv = sims[j]; best = j; }   // ties -> lowest index
      idx[i * TOPK + k] = best;
      sims[best] = -2e30f;
    }
  }
}

// x_lin = z_flat @ gnn_lin_W^T   (32768 x 16) @ (16 x 64)
__global__ void lin_x(const float* __restrict__ z, const float* __restrict__ Wl,
                      float* __restrict__ x_lin) {
  int id = blockIdx.x * blockDim.x + threadIdx.x;
  if (id >= BN_NODES * Dd) return;
  int n = id >> 6, dd = id & 63;
  const float* zr = z + n * 16;
  const float* wr = Wl + dd * 16;
  float s = 0.f;
#pragma unroll
  for (int k = 0; k < 16; ++k) s += zr[k] * wr[k];
  x_lin[id] = s;
}

// per-node attention scalars
__global__ void node_scores(const float* __restrict__ x_lin,
                            const float* __restrict__ embeds,
                            const float* __restrict__ att_i,
                            const float* __restrict__ att_j,
                            const float* __restrict__ att_em_i,
                            const float* __restrict__ att_em_j,
                            float* __restrict__ si, float* __restrict__ sj) {
  int n = blockIdx.x * blockDim.x + threadIdx.x;
  if (n >= BN_NODES) return;
  const float* xr = x_lin + (size_t)n * 64;
  const float* er = embeds + ((n & 127) << 6);
  float a = 0.f, b = 0.f;
  for (int d = 0; d < 64; ++d) {
    a += xr[d] * att_i[d] + er[d] * att_em_i[d];
    b += xr[d] * att_j[d] + er[d] * att_em_j[d];
  }
  si[n] = a; sj[n] = b;
}

// GAT softmax + aggregate. One block (64 threads) per destination node.
__global__ __launch_bounds__(64)
void gat_agg(const int* __restrict__ topk, const float* __restrict__ si,
             const float* __restrict__ sj, const float* __restrict__ x_lin,
             const float* __restrict__ gnn_bias, float* __restrict__ agg) {
  __shared__ float w[32];
  __shared__ int srcs[32];
  const int n = blockIdx.x, b = n >> 7, i = n & 127, t = threadIdx.x;
  if (t < 31) {
    int sN; bool valid;
    if (t < 30) { int j = topk[i * TOPK + t]; sN = (b << 7) | j; valid = (j != i); }
    else        { sN = n; valid = true; }
    srcs[t] = sN;
    float a = si[n] + sj[sN];
    a = (a > 0.f) ? a : 0.2f * a;              // leaky_relu(., 0.2)
    w[t] = valid ? a : -1e30f;
  }
  __syncthreads();
  if (t == 0) {
    float m = -3e38f;
    for (int k = 0; k < 31; ++k) m = fmaxf(m, w[k]);
    float ssum = 0.f;
    for (int k = 0; k < 31; ++k) {
      float e = (w[k] > -1e29f) ? __expf(w[k] - m) : 0.f;
      w[k] = e; ssum += e;
    }
    float inv = 1.f / ssum;
    for (int k = 0; k < 31; ++k) w[k] *= inv;
  }
  __syncthreads();
  float acc = gnn_bias[t];
  for (int k = 0; k < 31; ++k) acc += w[k] * x_lin[(size_t)srcs[k] * 64 + t];
  agg[(size_t)n * 64 + t] = acc;
}

// per-column stats over 32768 rows (64 columns). One block per column.
__global__ __launch_bounds__(256)
void col_stats64(const float* __restrict__ x, float* __restrict__ mean,
                 float* __restrict__ var) {
  __shared__ float s1[256], s2[256];
  const int c = blockIdx.x, t = threadIdx.x;
  float a = 0.f, b = 0.f;
  for (int r = t; r < BN_NODES; r += 256) {
    float v = x[(size_t)r * 64 + c];
    a += v; b += v * v;
  }
  s1[t] = a; s2[t] = b;
  __syncthreads();
  for (int off = 128; off > 0; off >>= 1) {
    if (t < off) { s1[t] += s1[t + off]; s2[t] += s2[t + off]; }
    __syncthreads();
  }
  if (t == 0) {
    float m = s1[0] * (1.f / (float)BN_NODES);
    mean[c] = m;
    var[c]  = s2[0] * (1.f / (float)BN_NODES) - m * m;
  }
}

// gout = relu(BN(agg)); t2 = gout * embeds  (in-place on agg)
__global__ void bn_relu_mulemb(float* __restrict__ agg,
                               const float* __restrict__ mean,
                               const float* __restrict__ var,
                               const float* __restrict__ g,
                               const float* __restrict__ be,
                               const float* __restrict__ embeds) {
  int id = blockIdx.x * blockDim.x + threadIdx.x;
  if (id >= BN_NODES * Dd) return;
  int c = id & 63, n = id >> 6;
  float v = agg[id];
  v = (v - mean[c]) * rsqrtf(var[c] + EPSf) * g[c] + be[c];
  v = fmaxf(v, 0.f);
  agg[id] = v * embeds[((n & 127) << 6) + c];
}

// x_frcst = relu(BN2(t2)) @ out_W + out_b
__global__ void forecast(const float* __restrict__ t2,
                         const float* __restrict__ mean,
                         const float* __restrict__ var,
                         const float* __restrict__ g,
                         const float* __restrict__ be,
                         const float* __restrict__ out_W,
                         const float* __restrict__ out_b,
                         float* __restrict__ x_frcst) {
  int n = blockIdx.x * blockDim.x + threadIdx.x;
  if (n >= BN_NODES) return;
  float s = out_b[0];
  for (int c = 0; c < 64; ++c) {
    float v = t2[(size_t)n * 64 + c];
    v = (v - mean[c]) * rsqrtf(var[c] + EPSf) * g[c] + be[c];
    v = fmaxf(v, 0.f);
    s += v * out_W[c];
  }
  x_frcst[n] = s;
}

__device__ __forceinline__ unsigned hash_u32(unsigned x) {
  x ^= x >> 17; x *= 0xed5ad4bbu; x ^= x >> 11; x *= 0xac4c1b51u;
  x ^= x >> 15; x *= 0x31848babu; x ^= x >> 14; return x;
}

// y_process = repeat(arange(4), B*S) as float; rhos = deterministic uniforms
__global__ void write_misc(float* __restrict__ yproc, float* __restrict__ rhos) {
  int id = blockIdx.x * blockDim.x + threadIdx.x;
  if (id < BN_NODES) yproc[id] = (float)(id >> 13);
  if (id < NWf) {
    unsigned h = hash_u32((unsigned)id * 2654435761u ^ 0x2a2a2a2au);
    rhos[id] = 1e-5f + ((float)h * (1.0f / 4294967296.0f)) * (0.01f - 1e-5f);
  }
}

// ---------------------------------------------------------------------------
extern "C" void kernel_launch(void* const* d_in, const int* in_sizes, int n_in,
                              void* d_out, int out_size, void* d_ws, size_t ws_size,
                              hipStream_t stream) {
  (void)in_sizes; (void)n_in; (void)out_size; (void)ws_size;

  const float* data       = (const float*)d_in[0];
  const float* enc_W      = (const float*)d_in[3];
  const float* enc_b      = (const float*)d_in[4];
  const float* enc_bn_g   = (const float*)d_in[5];
  const float* enc_bn_b   = (const float*)d_in[6];
  const float* dec_W      = (const float*)d_in[7];
  const float* dec_b      = (const float*)d_in[8];
  const float* dec_bn_g   = (const float*)d_in[9];
  const float* dec_bn_b   = (const float*)d_in[10];
  const float* emb_tables = (const float*)d_in[11];
  const float* gru_W_ih   = (const float*)d_in[12];
  const float* gru_b_ih   = (const float*)d_in[14];
  const float* gru_b_hh   = (const float*)d_in[15];
  const float* gnn_lin_W  = (const float*)d_in[16];
  const float* att_i      = (const float*)d_in[17];
  const float* att_j      = (const float*)d_in[18];
  const float* att_em_i   = (const float*)d_in[19];
  const float* att_em_j   = (const float*)d_in[20];
  const float* gnn_bias   = (const float*)d_in[21];
  const float* gnn_bn_g   = (const float*)d_in[22];
  const float* gnn_bn_b   = (const float*)d_in[23];
  const float* bn_out_g   = (const float*)d_in[24];
  const float* bn_out_b   = (const float*)d_in[25];
  const float* out_W      = (const float*)d_in[26];
  const float* out_b      = (const float*)d_in[27];

  // ---- output slots (concatenated flat, return order) ----
  float* o        = (float*)d_out;
  float* x_frcst  = o;                        // 32768
  float* x_recon  = x_frcst + 32768;          // 524288
  float* z_outp   = x_recon + 524288;         // 524288
  float* enc_fm   = z_outp + 524288;          // 524288
  float* dec_fm   = enc_fm + 524288;          // 524288
  float* all_emb  = dec_fm + 524288;          // 2097152
  float* y_proc   = all_emb + 2097152;        // 32768
  float* rhos     = y_proc + 32768;           // 2048
  float* rho_hat  = rhos + 2048;              // 2048

  // ---- workspace layout ----
  float* lin    = (float*)d_ws;               // 256*2048
  float* embeds = lin + 524288;               // 128*64
  int*   topk   = (int*)(embeds + 8192);      // 128*30
  float* xlin   = (float*)(topk + 3840);      // 32768*64
  float* si     = xlin + 2097152;             // 32768
  float* sj     = si + 32768;                 // 32768
  float* agg    = sj + 32768;                 // 32768*64
  float* cmean  = agg + 2097152;              // 64
  float* cvar   = cmean + 64;                 // 64
  float* cmean2 = cvar + 64;                  // 64
  float* cvar2  = cmean2 + 64;                // 64
  float* emean  = cvar2 + 64;                 // 2048
  float* evar   = emean + 2048;               // 2048

  const int gemmBlocks = (Bsz / BLK_M) * (NWf / BLK_N);   // 4*16 = 64
  const int elemBlocks = (Bsz * NWf) / 256;               // 2048

  // encoder layer 1
  gemm_bf16_wmma<<<gemmBlocks, 256, 0, stream>>>(data, enc_W, enc_b, lin,
                                                 Bsz, NWf, NWf);
  col_stats_gen<<<NWf / 256, 256, 0, stream>>>(lin, emean, evar, Bsz, NWf);
  bn_sigmoid_apply<<<elemBlocks, 256, 0, stream>>>(lin, emean, evar, enc_bn_g,
                                                   enc_bn_b, enc_fm, NWf - 1,
                                                   Bsz * NWf);
  // encoder layer 2 -> z
  gemm_bf16_wmma<<<gemmBlocks, 256, 0, stream>>>(enc_fm, enc_W + (size_t)NWf * NWf,
                                                 enc_b + NWf, lin, Bsz, NWf, NWf);
  col_stats_gen<<<NWf / 256, 256, 0, stream>>>(lin, emean, evar, Bsz, NWf);
  bn_sigmoid_apply<<<elemBlocks, 256, 0, stream>>>(lin, emean, evar,
                                                   enc_bn_g + NWf, enc_bn_b + NWf,
                                                   z_outp, NWf - 1, Bsz * NWf);
  // decoder layer 1
  gemm_bf16_wmma<<<gemmBlocks, 256, 0, stream>>>(z_outp, dec_W, dec_b, lin,
                                                 Bsz, NWf, NWf);
  col_stats_gen<<<NWf / 256, 256, 0, stream>>>(lin, emean, evar, Bsz, NWf);
  bn_sigmoid_apply<<<elemBlocks, 256, 0, stream>>>(lin, emean, evar, dec_bn_g,
                                                   dec_bn_b, dec_fm, NWf - 1,
                                                   Bsz * NWf);
  // decoder layer 2 -> x_recon
  gemm_bf16_wmma<<<gemmBlocks, 256, 0, stream>>>(dec_fm, dec_W + (size_t)NWf * NWf,
                                                 dec_b + NWf, lin, Bsz, NWf, NWf);
  col_stats_gen<<<NWf / 256, 256, 0, stream>>>(lin, emean, evar, Bsz, NWf);
  bn_sigmoid_apply<<<elemBlocks, 256, 0, stream>>>(lin, emean, evar,
                                                   dec_bn_g + NWf, dec_bn_b + NWf,
                                                   x_recon, NWf - 1, Bsz * NWf);

  // rho_hat = sum_batch(z)
  colsum<<<NWf / 256, 256, 0, stream>>>(z_outp, rho_hat, Bsz, NWf);

  // GRU sensor embeddings + all_emb tile
  gru_embed<<<Nn, 96, 0, stream>>>(emb_tables, gru_W_ih, gru_b_ih, gru_b_hh,
                                   embeds);
  write_allemb<<<(BN_NODES * Dd) / 256, 256, 0, stream>>>(embeds, all_emb);

  // dynamic top-k cosine graph
  topk_graph<<<Nn, 128, 0, stream>>>(embeds, topk);

  // GraphLayer
  lin_x<<<(BN_NODES * Dd) / 256, 256, 0, stream>>>(z_outp, gnn_lin_W, xlin);
  node_scores<<<BN_NODES / 256, 256, 0, stream>>>(xlin, embeds, att_i, att_j,
                                                  att_em_i, att_em_j, si, sj);
  gat_agg<<<BN_NODES, 64, 0, stream>>>(topk, si, sj, xlin, gnn_bias, agg);

  // GNN BN + relu, multiply by embeds, out BN + relu, forecast head
  col_stats64<<<Dd, 256, 0, stream>>>(agg, cmean, cvar);
  bn_relu_mulemb<<<(BN_NODES * Dd) / 256, 256, 0, stream>>>(agg, cmean, cvar,
                                                            gnn_bn_g, gnn_bn_b,
                                                            embeds);
  col_stats64<<<Dd, 256, 0, stream>>>(agg, cmean2, cvar2);
  forecast<<<BN_NODES / 256, 256, 0, stream>>>(agg, cmean2, cvar2, bn_out_g,
                                               bn_out_b, out_W, out_b, x_frcst);

  // misc outputs
  write_misc<<<BN_NODES / 256, 256, 0, stream>>>(y_proc, rhos);
}
